// DimeNet_5119601016936
// MI455X (gfx1250) — compile-verified, matrix operand order checked by reading, not run on hardware
//
#include <hip/hip_runtime.h>
#include <hip/hip_bf16.h>

// MI455X / gfx1250 DimeNet forward.
// All GEMMs have N==128 (H). Strategy: bf16 activations + v_wmma_f32_16x16x32_bf16,
// fused epilogues (bias+relu, gather-multiply, residual adds, fp32 atomic scatter-add).
// Workspace requirement: ~500 MB (padded bf16 activation buffers).

typedef __attribute__((ext_vector_type(16))) __bf16 v16bf;
typedef __attribute__((ext_vector_type(8)))  __bf16 v8bf;
typedef __attribute__((ext_vector_type(8)))  float  v8f;

// ---------------------------------------------------------------------------
// Generic WMMA GEMM: C[M,128] = relu(A[M,K] @ B[K,128] + bias) with optional
// epilogue: *mulG[gidx], +add1, +add2, atomic-scatter, f32/bf16 stores.
// Block = 256 threads (8 waves). Each wave owns a 16-row M-tile x full N=128.
// A is read directly from global (each row read exactly once since N==BN).
// B is staged in LDS transposed [n][k] with +8 element row pad (bank spread).
// ---------------------------------------------------------------------------
__launch_bounds__(256)
__global__ void wmma_gemm_n128(
    const __bf16* __restrict__ A, const __bf16* __restrict__ Bw,
    const float* __restrict__ bias,
    float* __restrict__ outF, __bf16* __restrict__ outB,
    const __bf16* __restrict__ mulG, const int* __restrict__ mulIdx,
    const float* __restrict__ add1, const float* __restrict__ add2,
    float* __restrict__ scat, const int* __restrict__ scatIdx,
    int M, int K)
{
  extern __shared__ __bf16 Bl[];                 // [128][K+8]
  const int ldsk = K + 8;
  for (int idx = threadIdx.x; idx < (K << 7); idx += 256) {
    int k = idx >> 7, n = idx & 127;
    Bl[n * ldsk + k] = Bw[idx];
  }
  __syncthreads();

  const int lane = threadIdx.x & 31;
  const int wave = threadIdx.x >> 5;
  const int lr = lane & 15;                      // row/col within 16
  const int lh = lane >> 4;                      // K-half selector
  const long row0 = (long)blockIdx.x * 128 + wave * 16;

  v8f acc[8] = {};
  // 16-bit A-frag layout: lanes 0-15 hold K {0..7,16..23}, lanes 16-31 {8..15,24..31}
  const __bf16* arow = A + (row0 + lr) * (long)K + lh * 8;

  for (int kb = 0; kb < K; kb += 32) {
    v8bf a0 = *(const v8bf*)(arow + kb);
    v8bf a1 = *(const v8bf*)(arow + kb + 16);
    v16bf af = __builtin_shufflevector(a0, a1, 0,1,2,3,4,5,6,7,8,9,10,11,12,13,14,15);
#pragma unroll
    for (int nt = 0; nt < 8; ++nt) {
      const __bf16* bp = Bl + (nt * 16 + lr) * ldsk + kb + lh * 8;
      v8bf b0 = *(const v8bf*)bp;
      v8bf b1 = *(const v8bf*)(bp + 16);
      v16bf bfv = __builtin_shufflevector(b0, b1, 0,1,2,3,4,5,6,7,8,9,10,11,12,13,14,15);
      acc[nt] = __builtin_amdgcn_wmma_f32_16x16x32_bf16(
          false, af, false, bfv, (short)0, acc[nt], false, false);
    }
  }

  // C layout: VGPR v -> row = row0 + v + 8*(lane>=16), col = nt*16 + lane%16
#pragma unroll
  for (int nt = 0; nt < 8; ++nt) {
    const int col = nt * 16 + lr;
    const float bv = bias ? bias[col] : 0.0f;
#pragma unroll
    for (int v = 0; v < 8; ++v) {
      const long row = row0 + v + (lh << 3);
      if (row >= M) continue;                    // pad rows discarded (buffers are M-padded)
      float val = acc[nt][v] + bv;
      val = fmaxf(val, 0.0f);
      if (mulG) {
        const long g = mulIdx ? (long)mulIdx[row] : row;
        val *= (float)mulG[g * 128 + col];
      }
      if (add1) val += add1[row * 128 + col];
      if (add2) val += add2[row * 128 + col];
      if (scat) atomicAdd(scat + (long)scatIdx[row] * 128 + col, val);
      if (outF) outF[row * 128 + col] = val;
      if (outB) outB[row * 128 + col] = (__bf16)val;
    }
  }
}

// ---------------------------------------------------------------------------
// Small helper kernels
// ---------------------------------------------------------------------------
__global__ void k_cvt_f32_bf16(const float* __restrict__ src, __bf16* __restrict__ dst, long n) {
  long i = (long)blockIdx.x * 256 + threadIdx.x;
  if (i < n) dst[i] = (__bf16)src[i];
}

// argmax(one-hot) -> embedding gather, bf16
__global__ void k_atoms(const float* __restrict__ atomf, const float* __restrict__ emb,
                        __bf16* __restrict__ xemb, int N) {
  int n = blockIdx.x;
  if (n >= N) return;
  __shared__ int s_t;
  if (threadIdx.x == 0) {
    const float* r = atomf + (long)n * 133;
    int best = 0; float bv = r[0];
    for (int c = 1; c < 100; ++c) { float v = r[c]; if (v > bv) { bv = v; best = c; } }
    s_t = best;
  }
  __syncthreads();
  xemb[(long)n * 128 + threadIdx.x] = (__bf16)emb[(long)s_t * 128 + threadIdx.x];
}

// Per-edge: rbf0 (Bessel*envelope), rbf_h = relu(rbf0@lin_rbf+b), build A0 and Aemb.
__global__ void k_edges(const float* __restrict__ atomf, const float* __restrict__ edgef,
                        const float* __restrict__ dist,
                        const float* __restrict__ lin_rbf_w, const float* __restrict__ lin_rbf_b,
                        const float* __restrict__ bessel,
                        const int* __restrict__ ii, const int* __restrict__ jj,
                        const __bf16* __restrict__ xemb,
                        __bf16* __restrict__ A0, __bf16* __restrict__ Aemb,
                        float* __restrict__ rbf0, int E) {
  const int e = blockIdx.x;
  const int t = threadIdx.x;            // 128 threads
  if (e >= gridDim.x) return;
  __shared__ float s_rbf[16];
  if (e >= (unsigned)E) {               // pad rows: zero-fill
    for (int c = t; c < 160; c += 128) A0[(long)e * 160 + c] = (__bf16)0.0f;
    for (int c = t; c < 384; c += 128) Aemb[(long)e * 384 + c] = (__bf16)0.0f;
    if (t < 16) rbf0[(long)e * 16 + t] = 0.0f;
    return;
  }
  if (t < 16) {
    float d = dist[e] * 0.125f;         // /CUTOFF
    float x5 = d * d; x5 = x5 * x5 * d; // d^5
    float env = (d < 1.0f)
        ? (1.0f / d - 28.0f * x5 + 48.0f * x5 * d - 21.0f * x5 * d * d) : 0.0f;
    float v = env * sinf(bessel[t] * d);
    s_rbf[t] = v;
    rbf0[(long)e * 16 + t] = v;
  }
  __syncthreads();
  float accv = lin_rbf_b[t];
  for (int k = 0; k < 16; ++k) accv += s_rbf[k] * lin_rbf_w[k * 128 + t];
  accv = fmaxf(accv, 0.0f);
  const int ie = ii[e], je = jj[e];
  __bf16* ar = Aemb + (long)e * 384;
  ar[t]       = xemb[(long)ie * 128 + t];
  ar[128 + t] = xemb[(long)je * 128 + t];
  ar[256 + t] = (__bf16)accv;
  // A0: [atom_feature[j] (133) || edge_feature (14) || zero pad to 160]
  __bf16* a0 = A0 + (long)e * 160;
  for (int c = t; c < 160; c += 128) {
    float v = 0.0f;
    if (c < 133)      v = atomf[(long)je * 133 + c];
    else if (c < 147) v = edgef[(long)e * 14 + (c - 133)];
    a0[c] = (__bf16)v;
  }
}

// sbf[t, l*16+r] = cos(l*angle[t]) * rbf0[idx_kj[t], r]
__global__ void k_sbf(const float* __restrict__ angle, const int* __restrict__ idx_kj,
                      const float* __restrict__ rbf0, __bf16* __restrict__ sbf, int T) {
  const int t = blockIdx.x;
  const int c = threadIdx.x;            // 96 threads
  if (t >= (unsigned)T) { sbf[(long)t * 96 + c] = (__bf16)0.0f; return; }
  const int l = c >> 4, r = c & 15;
  const float a = angle[t];
  const int e = idx_kj[t];
  sbf[(long)t * 96 + c] = (__bf16)(cosf(a * (float)l) * rbf0[(long)e * 16 + r]);
}

// atom_message[incomebond_index_to_atom[e]] += message[incomebond_edge_ids[e]]
__global__ void k_scatter_atoms(const float* __restrict__ msgF,
                                const int* __restrict__ eid, const int* __restrict__ ato,
                                float* __restrict__ atommsg, int E) {
  long idx = (long)blockIdx.x * 256 + threadIdx.x;
  long e = idx >> 7; int h = (int)(idx & 127);
  if (e < E)
    atomicAdd(atommsg + (long)ato[e] * 128 + h, msgF[(long)eid[e] * 128 + h]);
}

// AO = [atom_feature (133) || atom_message (128) || zero pad to 288]
__global__ void k_build_ao(const float* __restrict__ atomf, const float* __restrict__ atommsg,
                           __bf16* __restrict__ AO, int N) {
  const int n = blockIdx.x;
  for (int c = threadIdx.x; c < 288; c += 128) {
    float v = 0.0f;
    if (n < N) {
      if (c < 133)      v = atomf[(long)n * 133 + c];
      else if (c < 261) v = atommsg[(long)n * 128 + (c - 133)];
    }
    AO[(long)n * 288 + c] = (__bf16)v;
  }
}

// ---------------------------------------------------------------------------
// Host side
// ---------------------------------------------------------------------------
static inline void gemm(hipStream_t s, const __bf16* A, const __bf16* B, const float* bias,
                        float* outF, __bf16* outB, const __bf16* mulG, const int* mulIdx,
                        const float* add1, const float* add2,
                        float* scat, const int* scatIdx, int M, int K) {
  int grid = (M + 127) / 128;
  size_t shmem = (size_t)128 * (K + 8) * sizeof(__bf16);
  wmma_gemm_n128<<<grid, 256, shmem, s>>>(A, B, bias, outF, outB, mulG, mulIdx,
                                          add1, add2, scat, scatIdx, M, K);
}

static inline void cvt(hipStream_t s, const float* src, __bf16* dst, long n) {
  k_cvt_f32_bf16<<<(int)((n + 255) / 256), 256, 0, s>>>(src, dst, n);
}

extern "C" void kernel_launch(void* const* d_in, const int* in_sizes, int n_in,
                              void* d_out, int out_size, void* d_ws, size_t ws_size,
                              hipStream_t stream) {
  (void)n_in; (void)out_size; (void)ws_size;
  // ----- inputs -----
  const float* atom_feature = (const float*)d_in[0];
  const float* edge_feature = (const float*)d_in[1];
  const float* dist         = (const float*)d_in[2];
  const float* angle        = (const float*)d_in[3];
  const float* W_i1_w   = (const float*)d_in[4];
  const float* W_i1_b   = (const float*)d_in[5];
  const float* emb_tab  = (const float*)d_in[6];
  const float* lin_rbf_w = (const float*)d_in[7];
  const float* lin_rbf_b = (const float*)d_in[8];
  const float* lin_emb_w = (const float*)d_in[9];
  const float* lin_emb_b = (const float*)d_in[10];
  const float* bessel    = (const float*)d_in[11];
  const float* L_rbf2_w = (const float*)d_in[12];
  const float* L_rbf2_b = (const float*)d_in[13];
  const float* L_kj_w   = (const float*)d_in[14];
  const float* L_kj_b   = (const float*)d_in[15];
  const float* L_sbf1_w = (const float*)d_in[16];
  const float* L_sbf2_w = (const float*)d_in[17];
  const float* L_down_w = (const float*)d_in[18];
  const float* L_down_b = (const float*)d_in[19];
  const float* L_up_w   = (const float*)d_in[20];
  const float* L_up_b   = (const float*)d_in[21];
  const float* L_res1_w = (const float*)d_in[22];
  const float* L_res1_b = (const float*)d_in[23];
  const float* L_res2_w = (const float*)d_in[24];
  const float* L_res2_b = (const float*)d_in[25];
  const float* W_o_w = (const float*)d_in[26];
  const float* W_o_b = (const float*)d_in[27];
  const int* idx_i  = (const int*)d_in[28];
  const int* idx_j  = (const int*)d_in[29];
  const int* idx_kj = (const int*)d_in[30];
  const int* ib_eid = (const int*)d_in[32];
  const int* ib_ato = (const int*)d_in[33];

  const int N = in_sizes[0] / 133;
  const int E = in_sizes[2];
  const int T = in_sizes[3];
  const long Npad = ((N + 127) / 128) * 128L;
  const long Epad = ((E + 127) / 128) * 128L;
  const long Tpad = ((T + 127) / 128) * 128L;

  // ----- workspace layout -----
  size_t off = 0;
  auto alloc = [&](size_t bytes) -> char* {
    char* p = (char*)d_ws + off;
    off += (bytes + 255) & ~(size_t)255;
    return p;
  };
  float* msgF    = (float*)alloc(Epad * 128 * 4);
  float* aggF    = (float*)alloc(Epad * 128 * 4);
  float* atommsg = (float*)alloc(Npad * 128 * 4);
  float* rbf0    = (float*)alloc(Epad * 16 * 4);
  __bf16* msgB   = (__bf16*)alloc(Epad * 128 * 2);
  __bf16* rbfeB  = (__bf16*)alloc(Epad * 128 * 2);
  __bf16* xkjB   = (__bf16*)alloc(Epad * 128 * 2);
  __bf16* aggB   = (__bf16*)alloc(Epad * 128 * 2);
  __bf16* xembB  = (__bf16*)alloc(Npad * 128 * 2);
  // region1: A0 [Epad,160] (used only before sbf is built) aliases sbf [Tpad,96]
  char* region1 = alloc((size_t)Tpad * 96 * 2);
  __bf16* A0B  = (__bf16*)region1;
  __bf16* sbfB = (__bf16*)region1;
  // region2: Aemb [Epad,384] (used only before layer loop) aliases s1 [Tpad,128]
  char* region2 = alloc((size_t)Tpad * 128 * 2);
  __bf16* AembB = (__bf16*)region2;
  __bf16* s1B   = (__bf16*)region2;
  __bf16* AOB   = (__bf16*)alloc(Npad * 288 * 2);
  // bf16 weights
  __bf16* Wi1B   = (__bf16*)alloc(160 * 128 * 2);
  __bf16* LembB  = (__bf16*)alloc(384 * 128 * 2);
  __bf16* WoB    = (__bf16*)alloc(288 * 128 * 2);
  __bf16* kjB[2], *rbf2B[2], *downB[2], *sbf1B[2], *sbf2B[2], *upB[2], *res1B[2], *res2B[2];
  for (int l = 0; l < 2; ++l) {
    kjB[l]   = (__bf16*)alloc(128 * 128 * 2);
    rbf2B[l] = (__bf16*)alloc(128 * 128 * 2);
    downB[l] = (__bf16*)alloc(128 * 128 * 2);
    sbf1B[l] = (__bf16*)alloc(96 * 128 * 2);
    sbf2B[l] = (__bf16*)alloc(128 * 128 * 2);
    upB[l]   = (__bf16*)alloc(128 * 128 * 2);
    res1B[l] = (__bf16*)alloc(128 * 128 * 2);
    res2B[l] = (__bf16*)alloc(128 * 128 * 2);
  }

  // ----- weight conversion (f32 -> bf16, zero-padded K) -----
  hipMemsetAsync(Wi1B, 0, 160 * 128 * 2, stream);
  cvt(stream, W_i1_w, Wi1B, 147L * 128);
  cvt(stream, lin_emb_w, LembB, 384L * 128);
  hipMemsetAsync(WoB, 0, 288 * 128 * 2, stream);
  cvt(stream, W_o_w, WoB, 261L * 128);
  for (int l = 0; l < 2; ++l) {
    cvt(stream, L_kj_w   + l * 128 * 128, kjB[l],   128L * 128);
    cvt(stream, L_rbf2_w + l * 128 * 128, rbf2B[l], 128L * 128);
    cvt(stream, L_down_w + l * 128 * 128, downB[l], 128L * 128);
    cvt(stream, L_sbf1_w + l * 96 * 128,  sbf1B[l],  96L * 128);
    cvt(stream, L_sbf2_w + l * 128 * 128, sbf2B[l], 128L * 128);
    cvt(stream, L_up_w   + l * 128 * 128, upB[l],   128L * 128);
    cvt(stream, L_res1_w + l * 128 * 128, res1B[l], 128L * 128);
    cvt(stream, L_res2_w + l * 128 * 128, res2B[l], 128L * 128);
  }

  // ----- prep -----
  hipMemsetAsync(atommsg, 0, Npad * 128 * 4, stream);
  k_atoms<<<N, 128, 0, stream>>>(atom_feature, emb_tab, xembB, N);
  k_edges<<<(int)Epad, 128, 0, stream>>>(atom_feature, edge_feature, dist,
                                         lin_rbf_w, lin_rbf_b, bessel,
                                         idx_i, idx_j, xembB, A0B, AembB, rbf0, E);

  // initial message = relu([x_j || edge_feat] @ W_i1 + b)     (K=160 padded)
  gemm(stream, A0B, Wi1B, W_i1_b, msgF, msgB,
       nullptr, nullptr, nullptr, nullptr, nullptr, nullptr, E, 160);
  // rbf_e = relu([x_emb_i || x_emb_j || rbf_h] @ lin_emb + b) (K=384)
  gemm(stream, AembB, LembB, lin_emb_b, nullptr, rbfeB,
       nullptr, nullptr, nullptr, nullptr, nullptr, nullptr, E, 384);
  // sbf (overwrites the A0 region — safe, initial GEMM already consumed it)
  k_sbf<<<(int)Tpad, 96, 0, stream>>>(angle, idx_kj, rbf0, sbfB, T);

  // ----- interaction layers -----
  for (int l = 0; l < 2; ++l) {
    // x_kj = relu(msg @ W_kj + b)
    gemm(stream, msgB, kjB[l], L_kj_b + l * 128, nullptr, xkjB,
         nullptr, nullptr, nullptr, nullptr, nullptr, nullptr, E, 128);
    // x_kj *= relu(rbf_e @ W_rbf2 + b)   (fused multiply, in-place into xkjB)
    gemm(stream, rbfeB, rbf2B[l], L_rbf2_b + l * 128, nullptr, xkjB,
         xkjB, nullptr, nullptr, nullptr, nullptr, nullptr, E, 128);
    // x_kj = relu(x_kj @ W_down + b)     (in-place: each wave owns its rows)
    gemm(stream, xkjB, downB[l], L_down_b + l * 128, nullptr, xkjB,
         nullptr, nullptr, nullptr, nullptr, nullptr, nullptr, E, 128);
    // s = relu(sbf @ W_sbf1)             (K=96; s1 region overwrites Aemb)
    gemm(stream, sbfB, sbf1B[l], nullptr, nullptr, s1B,
         nullptr, nullptr, nullptr, nullptr, nullptr, nullptr, T, 96);
    // a = relu(s @ W_sbf2) * x_kj[idx_kj]  (fused gather-multiply, in-place)
    gemm(stream, s1B, sbf2B[l], nullptr, nullptr, s1B,
         xkjB, idx_kj, nullptr, nullptr, nullptr, nullptr, T, 128);
    // agg[idx_kj] += relu(a @ W_up + b)  (fused fp32 atomic scatter; agg fits in L2)
    hipMemsetAsync(aggF, 0, Epad * 128 * 4, stream);
    gemm(stream, s1B, upB[l], L_up_b + l * 128, nullptr, nullptr,
         nullptr, nullptr, nullptr, nullptr, aggF, idx_kj, T, 128);
    cvt(stream, aggF, aggB, Epad * 128);
    // h = relu(agg @ W_res1 + b)         (in-place into aggB; aggF keeps the value)
    gemm(stream, aggB, res1B[l], L_res1_b + l * 128, nullptr, aggB,
         nullptr, nullptr, nullptr, nullptr, nullptr, nullptr, E, 128);
    // message = relu(h @ W_res2 + b) + agg + message
    gemm(stream, aggB, res2B[l], L_res2_b + l * 128, msgF, msgB,
         nullptr, nullptr, aggF, msgF, nullptr, nullptr, E, 128);
  }

  // ----- readout -----
  k_scatter_atoms<<<(int)(((long)E * 128 + 255) / 256), 256, 0, stream>>>(
      msgF, ib_eid, ib_ato, atommsg, E);
  k_build_ao<<<(int)Npad, 128, 0, stream>>>(atom_feature, atommsg, AOB, N);
  // out = relu([atom_feature || atom_message] @ W_o + b)   (K=288 padded)
  gemm(stream, AOB, WoB, W_o_b, (float*)d_out, nullptr,
       nullptr, nullptr, nullptr, nullptr, nullptr, nullptr, N, 288);
}
